// GIN_56831007261128
// MI455X (gfx1250) — matrix operand order repaired and checked by reading, not compiled, not run on hardware
//
#include <hip/hip_runtime.h>
#include <hip/hip_bf16.h>

typedef __attribute__((ext_vector_type(16))) _Float16 v16h;
typedef __attribute__((ext_vector_type(8)))  _Float16 v8h;
typedef __attribute__((ext_vector_type(4)))  _Float16 v4h;
typedef __attribute__((ext_vector_type(8)))  float    v8f;

constexpr int N = 100000;
constexpr int E = 600000;
constexpr int H = 128;
constexpr int G = 1000;
constexpr int LDH = 136;          // padded halves per LDS row (272B, 16B-aligned)

// CDNA5 packed f16 atomic add (resolved at L2) — no portable builtin on gfx1250.
__device__ __forceinline__ void atomic_pk_add_f16(_Float16* addr, unsigned packed) {
    asm volatile("global_atomic_pk_add_f16 %0, %1, off"
                 :: "v"(addr), "v"(packed) : "memory");
}

// ---------------- prep: fold BN constants -------------------------------
__global__ void gin_prep_bn(const float* __restrict__ g0, const float* __restrict__ be0,
                            const float* __restrict__ m0, const float* __restrict__ v0,
                            const float* __restrict__ b10, const float* __restrict__ b20,
                            const float* __restrict__ gs, const float* __restrict__ bes,
                            const float* __restrict__ ms, const float* __restrict__ vs,
                            const float* __restrict__ b1s, const float* __restrict__ b2s,
                            float* __restrict__ bnA, float* __restrict__ bnB,
                            float* __restrict__ b2c) {
    int i = blockIdx.x * blockDim.x + threadIdx.x;
    if (i >= 5 * H) return;
    int l = i / H, n = i % H;
    float g, be, m, v, b1, b2;
    if (l == 0) { g = g0[n]; be = be0[n]; m = m0[n]; v = v0[n]; b1 = b10[n]; b2 = b20[n]; }
    else { int o = (l - 1) * H + n; g = gs[o]; be = bes[o]; m = ms[o]; v = vs[o]; b1 = b1s[o]; b2 = b2s[o]; }
    float A = g * rsqrtf(v + 1e-5f);
    bnA[i] = A;
    bnB[i] = (b1 - m) * A + be;
    b2c[i] = b2;
}

// ------------- prep: pack 9 weight matrices into WMMA B-fragment layout --
// Fragment layout (16x16x32 f16 B operand): lane l covers column n = (l&15),
// K kb..kb+7 then kb+16..kb+23 where kb = kt*32 + ((l>=16)?8:0).
__global__ void gin_pack_w(const float* __restrict__ W2_0,
                           const float* __restrict__ W1s,
                           const float* __restrict__ W2s,
                           _Float16* __restrict__ out) {
    int idx = blockIdx.x * blockDim.x + threadIdx.x;
    if (idx >= 9 * 4 * 8 * 32) return;
    int lane = idx & 31;
    int nt   = (idx >> 5) & 7;
    int kt   = (idx >> 8) & 3;
    int mat  = idx >> 10;
    const float* W;
    if (mat == 0)      W = W2_0;
    else if (mat <= 4) W = W1s + (size_t)(mat - 1) * H * H;
    else               W = W2s + (size_t)(mat - 5) * H * H;
    int n  = nt * 16 + (lane & 15);
    int kb = kt * 32 + ((lane >> 4) ? 8 : 0);
    _Float16* o = out + (((size_t)(mat * 4 + kt) * 8 + nt) * 32 + lane) * 16;
#pragma unroll
    for (int j = 0; j < 16; ++j) o[j] = (_Float16)W[(size_t)(kb + j + (j >= 8 ? 8 : 0)) * H + n];
}

// ---------------- layer-0 edge aggregation (3 features, f32) --------------
__global__ void gin_agg3(const float* __restrict__ x, const int* __restrict__ ei,
                         float* __restrict__ agg3) {
    int e = blockIdx.x * blockDim.x + threadIdx.x;
    if (e >= E) return;
    int src = ei[e], dst = ei[E + e];
    float a = x[(size_t)src * 3 + 0];
    float b = x[(size_t)src * 3 + 1];
    float c = x[(size_t)src * 3 + 2];
    atomicAdd(&agg3[(size_t)dst * 3 + 0], a);
    atomicAdd(&agg3[(size_t)dst * 3 + 1], b);
    atomicAdd(&agg3[(size_t)dst * 3 + 2], c);
}

// ------------- layer-0 front: (x+agg)@W1_0 (K=3) + BN + ReLU -> f16 h ----
__global__ void gin_l0_front(const float* __restrict__ x, const float* __restrict__ agg3,
                             const float* __restrict__ W1_0,
                             const float* __restrict__ bnA, const float* __restrict__ bnB,
                             _Float16* __restrict__ h1) {
    long i = (long)blockIdx.x * blockDim.x + threadIdx.x;
    if (i >= (long)N * H) return;
    int node = (int)(i >> 7);
    int n = (int)(i & 127);
    float t0 = x[(size_t)node * 3 + 0] + agg3[(size_t)node * 3 + 0];
    float t1 = x[(size_t)node * 3 + 1] + agg3[(size_t)node * 3 + 1];
    float t2 = x[(size_t)node * 3 + 2] + agg3[(size_t)node * 3 + 2];
    float d = t0 * W1_0[n] + t1 * W1_0[H + n] + t2 * W1_0[2 * H + n];
    float v = d * bnA[n] + bnB[n];
    h1[(size_t)node * H + n] = (_Float16)(v > 0.f ? v : 0.f);
}

// ----- edge aggregation (128 f16 features, packed-f16 atomics) ------------
__global__ void gin_agg128(const _Float16* __restrict__ h, const int* __restrict__ ei,
                           _Float16* __restrict__ agg) {
    long idx = (long)blockIdx.x * blockDim.x + threadIdx.x;
    int e = (int)(idx >> 5);
    if (e >= E) return;
    int lane = (int)(idx & 31);
    int src = ei[e], dst = ei[E + e];
    const uint2 v = *(const uint2*)(h + (size_t)src * H + lane * 4);
    _Float16* a = agg + (size_t)dst * H + lane * 4;
    atomic_pk_add_f16(a + 0, v.x);
    atomic_pk_add_f16(a + 2, v.y);
}

// ---------------- fused MLP: 128 rows/block, 8 waves ---------------------
// doG1==1: stage f16(h+agg) -> GEMM1(W1) -> BN+ReLU -> LDS -> GEMM2(W2)+b2+ReLU
// doG1==0: stage f16 input directly -> GEMM2(W2)+b2+ReLU
// Output written through LDS for coalesced b128 stores. In-place (hOut==hIn) safe:
// each block reads only its own rows, before writing them.
__global__ __launch_bounds__(256) void gin_fused_mlp(
    const _Float16* __restrict__ hIn, const _Float16* __restrict__ aggIn,
    const _Float16* __restrict__ w1p, const _Float16* __restrict__ w2p,
    const float* __restrict__ bnA, const float* __restrict__ bnB,
    const float* __restrict__ b2, _Float16* __restrict__ hOut,
    int nRows, int doG1) {
    __shared__ __align__(16) _Float16 sT[128 * LDH];   // 34.8 KB
    const int tid  = threadIdx.x;
    const int lane = tid & 31;
    const int wave = tid >> 5;
    const int rowBase0 = blockIdx.x * 128;

    const int mRow     = wave * 16 + (lane & 15);
    const int kHalfSel = (lane >> 4) * 8;   // A-frag K sub-offset per lane half
    const int n0       = lane & 15;         // C/D column within tile
    const int mB       = (lane >> 4) * 8;   // C/D row offset per lane half

    // stage input rows into LDS (f16, coalesced b128)
    for (int i = tid; i < 128 * (H / 8); i += 256) {
        int r = i >> 4;
        int c = (i & 15) * 8;
        int gr = rowBase0 + r;
        v8h v = {};
        if (gr < nRows) {
            v = *(const v8h*)(hIn + (size_t)gr * H + c);
            if (doG1) {
                v8h a = *(const v8h*)(aggIn + (size_t)gr * H + c);
                v = v + a;            // v_pk_add_f16
            }
        }
        *(v8h*)(&sT[r * LDH + c]) = v;
    }
    __syncthreads();

    if (doG1) {
        // GEMM1: this wave's 16 rows x 128 cols
        v8f acc[8];
        v8f zero = {};
#pragma unroll
        for (int nt = 0; nt < 8; ++nt) acc[nt] = zero;
#pragma unroll
        for (int kt = 0; kt < 4; ++kt) {
            int k0 = kt * 32 + kHalfSel;
            v8h lo = *(const v8h*)(&sT[mRow * LDH + k0]);
            v8h hi = *(const v8h*)(&sT[mRow * LDH + k0 + 16]);
            v16h af;
#pragma unroll
            for (int i = 0; i < 8; ++i) { af[i] = lo[i]; af[i + 8] = hi[i]; }
#pragma unroll
            for (int nt = 0; nt < 8; ++nt) {
                v16h bf = *(const v16h*)(w1p + (((size_t)(kt * 8 + nt)) * 32 + lane) * 16);
                acc[nt] = __builtin_amdgcn_wmma_f32_16x16x32_f16(
                    false, af, false, bf, (short)0, acc[nt], false, false);
            }
        }
        // BN + ReLU, write back into this wave's own 16 LDS rows (f16)
#pragma unroll
        for (int nt = 0; nt < 8; ++nt) {
            int n = nt * 16 + n0;
            float A = bnA[n], B = bnB[n];
#pragma unroll
            for (int r = 0; r < 8; ++r) {
                float v = acc[nt][r] * A + B;
                v = v > 0.f ? v : 0.f;
                sT[(wave * 16 + mB + r) * LDH + n] = (_Float16)v;
            }
        }
    }

    // GEMM2 (+ b2, ReLU)
    v8f acc2[8];
    v8f zero2 = {};
#pragma unroll
    for (int nt = 0; nt < 8; ++nt) acc2[nt] = zero2;
#pragma unroll
    for (int kt = 0; kt < 4; ++kt) {
        int k0 = kt * 32 + kHalfSel;
        v8h lo = *(const v8h*)(&sT[mRow * LDH + k0]);
        v8h hi = *(const v8h*)(&sT[mRow * LDH + k0 + 16]);
        v16h af;
#pragma unroll
        for (int i = 0; i < 8; ++i) { af[i] = lo[i]; af[i + 8] = hi[i]; }
#pragma unroll
        for (int nt = 0; nt < 8; ++nt) {
            v16h bf = *(const v16h*)(w2p + (((size_t)(kt * 8 + nt)) * 32 + lane) * 16);
            acc2[nt] = __builtin_amdgcn_wmma_f32_16x16x32_f16(
                false, af, false, bf, (short)0, acc2[nt], false, false);
        }
    }
    // write results into own LDS rows (f16)
#pragma unroll
    for (int nt = 0; nt < 8; ++nt) {
        int n = nt * 16 + n0;
        float bb = b2[n];
#pragma unroll
        for (int r = 0; r < 8; ++r) {
            float v = acc2[nt][r] + bb;
            v = v > 0.f ? v : 0.f;
            sT[(wave * 16 + mB + r) * LDH + n] = (_Float16)v;
        }
    }
    __syncthreads();
    // coalesced b128 copy-out
    for (int i = tid; i < 128 * (H / 8); i += 256) {
        int r = i >> 4;
        int c = (i & 15) * 8;
        int gr = rowBase0 + r;
        if (gr < nRows)
            *(v8h*)(hOut + (size_t)gr * H + c) = *(const v8h*)(&sT[r * LDH + c]);
    }
}

// ---------------- pooling ------------------------------------------------
__global__ void gin_pool_init(float* __restrict__ out, const float* __restrict__ bc) {
    int g = blockIdx.x * blockDim.x + threadIdx.x;
    if (g < G) out[g] = bc[0];
}

__global__ void gin_pool(const _Float16* __restrict__ h, const float* __restrict__ Wc,
                         const int* __restrict__ batch, float* __restrict__ out) {
    long idx = (long)blockIdx.x * blockDim.x + threadIdx.x;
    int node = (int)(idx >> 5);
    if (node >= N) return;
    int lane = (int)(idx & 31);
    const v4h hv = *(const v4h*)(h + (size_t)node * H + lane * 4);
    const float4 wv = *(const float4*)(Wc + lane * 4);
    float s = (float)hv[0] * wv.x + (float)hv[1] * wv.y +
              (float)hv[2] * wv.z + (float)hv[3] * wv.w;
#pragma unroll
    for (int off = 16; off >= 1; off >>= 1) s += __shfl_xor(s, off);
    if (lane == 0) atomicAdd(&out[batch[node]], s);
}

// ---------------- host launch --------------------------------------------
extern "C" void kernel_launch(void* const* d_in, const int* in_sizes, int n_in,
                              void* d_out, int out_size, void* d_ws, size_t ws_size,
                              hipStream_t stream) {
    const float* x     = (const float*)d_in[0];
    const int*   ei    = (const int*)d_in[1];
    const int*   batch = (const int*)d_in[2];
    const float* W1_0  = (const float*)d_in[3];
    const float* b1_0  = (const float*)d_in[4];
    const float* g_0   = (const float*)d_in[5];
    const float* be_0  = (const float*)d_in[6];
    const float* m_0   = (const float*)d_in[7];
    const float* v_0   = (const float*)d_in[8];
    const float* W2_0  = (const float*)d_in[9];
    const float* b2_0  = (const float*)d_in[10];
    const float* W1s   = (const float*)d_in[11];
    const float* b1s   = (const float*)d_in[12];
    const float* gs    = (const float*)d_in[13];
    const float* bes   = (const float*)d_in[14];
    const float* ms    = (const float*)d_in[15];
    const float* vs    = (const float*)d_in[16];
    const float* W2s   = (const float*)d_in[17];
    const float* b2s   = (const float*)d_in[18];
    const float* Wc    = (const float*)d_in[19];
    const float* bc    = (const float*)d_in[20];
    float* out = (float*)d_out;

    char* ws = (char*)d_ws;
    const size_t hBytes = (size_t)N * H * sizeof(_Float16);     // 25.6 MB
    _Float16* h    = (_Float16*)ws;
    _Float16* agg  = (_Float16*)(ws + hBytes);                  // 25.6 MB
    float*    agg3 = (float*)(ws + 2 * hBytes);                 // 1.2 MB
    const size_t agg3Bytes = (size_t)N * 3 * sizeof(float);
    _Float16* wpack = (_Float16*)(ws + 2 * hBytes + ((agg3Bytes + 255) & ~(size_t)255));
    const size_t wpackBytes = (size_t)9 * H * H * sizeof(_Float16);
    float* bnA = (float*)((char*)wpack + wpackBytes);
    float* bnB = bnA + 5 * H;
    float* b2c = bnB + 5 * H;

    gin_prep_bn<<<(5 * H + 255) / 256, 256, 0, stream>>>(
        g_0, be_0, m_0, v_0, b1_0, b2_0, gs, bes, ms, vs, b1s, b2s, bnA, bnB, b2c);
    gin_pack_w<<<(9 * 4 * 8 * 32 + 255) / 256, 256, 0, stream>>>(W2_0, W1s, W2s, wpack);

    // layer 0
    hipMemsetAsync(agg3, 0, agg3Bytes, stream);
    gin_agg3<<<(E + 255) / 256, 256, 0, stream>>>(x, ei, agg3);
    gin_l0_front<<<(unsigned)(((long)N * H + 255) / 256), 256, 0, stream>>>(
        x, agg3, W1_0, bnA, bnB, h);
    gin_fused_mlp<<<(N + 127) / 128, 256, 0, stream>>>(
        h, nullptr, nullptr, wpack /*mat 0 = W2_0*/,
        nullptr, nullptr, b2c, h, N, 0);

    // layers 1..4
    for (int l = 0; l < 4; ++l) {
        hipMemsetAsync(agg, 0, hBytes, stream);
        gin_agg128<<<(unsigned)(((long)E * 32 + 255) / 256), 256, 0, stream>>>(h, ei, agg);
        const _Float16* w1p = wpack + (size_t)(1 + l) * H * H;
        const _Float16* w2p = wpack + (size_t)(5 + l) * H * H;
        gin_fused_mlp<<<(N + 127) / 128, 256, 0, stream>>>(
            h, agg, w1p, w2p,
            bnA + (1 + l) * H, bnB + (1 + l) * H, b2c + (1 + l) * H, h, N, 1);
    }

    // pool + classifier
    gin_pool_init<<<(G + 255) / 256, 256, 0, stream>>>(out, bc);
    gin_pool<<<(unsigned)(((long)N * 32 + 255) / 256), 256, 0, stream>>>(h, Wc, batch, out);
}
